// GraphTransformerLayer_60662118089125
// MI455X (gfx1250) — matrix operand order repaired and checked by reading, not compiled, not run on hardware
//
#include <hip/hip_runtime.h>
#include <hip/hip_bf16.h>
#include <stdint.h>

typedef __attribute__((ext_vector_type(16))) __bf16 v16bf;
typedef __attribute__((ext_vector_type(8)))  float  v8f;

#ifndef ATHENA_NO_ASYNC
#define USE_ASYNC 1
#else
#define USE_ASYNC 0
#endif

__device__ __forceinline__ unsigned short f32_to_bf16_rte(float f) {
  union { float f; unsigned u; } x; x.f = f;
  unsigned r = x.u + 0x7fffu + ((x.u >> 16) & 1u);
  return (unsigned short)(r >> 16);
}

// pack two fp32 -> one dword holding two RTE-rounded bf16 (lo = a, hi = b)
__device__ __forceinline__ unsigned pack2_bf16_rte(float a, float b) {
  union { float f; unsigned u; } xa, xb; xa.f = a; xb.f = b;
  unsigned ra = xa.u + 0x7fffu + ((xa.u >> 16) & 1u);
  unsigned rb = xb.u + 0x7fffu + ((xb.u >> 16) & 1u);
  return (ra >> 16) | (rb & 0xffff0000u);
}

union Frag16 { uint4 q[2]; v16bf v; };

__device__ __forceinline__ v8f wmma_bf16(const Frag16& a, const Frag16& b, v8f c) {
  return __builtin_amdgcn_wmma_f32_16x16x32_bf16(false, a.v, false, b.v, (short)0, c,
                                                 false, false);
}

__device__ __forceinline__ v8f v8f_zero() {
  v8f z;
#pragma unroll
  for (int i = 0; i < 8; ++i) z[i] = 0.f;
  return z;
}

// ---------------------------------------------------------------------------
// Generic bf16 WMMA GEMM:  C[M,N] = act(A[M,K] @ Bt[N,K]^T + bias)
// A row-major bf16, Bt is B stored N-major (Bt[n*K+k] = B[k][n]) bf16.
// Block = 256 threads = 8 waves (2 x 4); block tile 128x128; wave tile 64x32.
// A/B fragments loaded straight from global (L2-resident operands).
// ---------------------------------------------------------------------------
template <int RELU>
__global__ __launch_bounds__(256) void gemm_bf16(
    const unsigned short* __restrict__ A,
    const unsigned short* __restrict__ Bt,
    const float* __restrict__ bias,
    float* __restrict__ Cf,            // optional fp32 output
    unsigned short* __restrict__ Cb,   // optional bf16 output
    int M, int N, int K)
{
  const int tid  = threadIdx.x;
  const int wave = tid >> 5, lane = tid & 31;
  const int wy = wave >> 2, wx = wave & 3;
  const int l16 = lane & 15, hi = lane >> 4;
  const int m0 = blockIdx.y * 128 + wy * 64;
  const int n0 = blockIdx.x * 128 + wx * 32;

  v8f acc[4][2];
#pragma unroll
  for (int mt = 0; mt < 4; ++mt)
#pragma unroll
    for (int nt = 0; nt < 2; ++nt) acc[mt][nt] = v8f_zero();

  for (int kb = 0; kb < K; kb += 32) {
    if (kb + 32 < K) {  // global_prefetch_b8 of next K-slab
      __builtin_prefetch(A + (size_t)(m0 + l16) * K + kb + 32, 0, 1);
      __builtin_prefetch(Bt + (size_t)(n0 + l16) * K + kb + 32, 0, 1);
    }
    Frag16 bF[2];
#pragma unroll
    for (int nt = 0; nt < 2; ++nt) {
      // B 32x16 bf16 frag: lane holds col n, 16 consecutive K (lo/hi half-wave)
      const unsigned short* p = Bt + (size_t)(n0 + nt * 16 + l16) * K + kb + hi * 16;
      bF[nt].q[0] = *(const uint4*)p;
      bF[nt].q[1] = *(const uint4*)(p + 8);
    }
#pragma unroll
    for (int mt = 0; mt < 4; ++mt) {
      // A 16x32 bf16 frag: lane holds row m, 16 consecutive K (lo/hi half-wave)
      Frag16 aF;
      const unsigned short* p = A + (size_t)(m0 + mt * 16 + l16) * K + kb + hi * 16;
      aF.q[0] = *(const uint4*)p;
      aF.q[1] = *(const uint4*)(p + 8);
      acc[mt][0] = wmma_bf16(aF, bF[0], acc[mt][0]);
      acc[mt][1] = wmma_bf16(aF, bF[1], acc[mt][1]);
    }
  }

#pragma unroll
  for (int nt = 0; nt < 2; ++nt) {
    const int n = n0 + nt * 16 + l16;
    const float bs = bias ? bias[n] : 0.f;
#pragma unroll
    for (int mt = 0; mt < 4; ++mt) {
#pragma unroll
      for (int p = 0; p < 8; ++p) {
        const int m = m0 + mt * 16 + p + hi * 8;  // C layout: vgpr p -> row
        float v = acc[mt][nt][p] + bs;
        if (RELU) v = v > 0.f ? v : 0.f;
        const size_t idx = (size_t)m * N + n;
        if (Cf) Cf[idx] = v;
        if (Cb) Cb[idx] = f32_to_bf16_rte(v);
      }
    }
  }
}

// ---------------------------------------------------------------------------
// Fused relation-bias attention. One block per (i, b); wave == head.
// Per block:  ra/rb = relation[j,i,b,:] @ rel_w  (bf16 WMMA, K=512),
//             w[j,h] = scl^2 * sum_n (q+ra)(k+rb)  (shuffle reduce),
//             softmax over j, attn = P @ V.  Output bf16.
// relation tile is double-buffer pipelined into LDS with
// global_load_async_to_lds_b128 (ASYNCcnt), then converted fp32->bf16 in LDS
// with packed dword stores (2 bf16 per ds_store_b32).
// ---------------------------------------------------------------------------
__global__ __launch_bounds__(256) void rel_attn(
    const float* __restrict__ rel,            // (T,T,B,512)
    const unsigned short* __restrict__ relWt, // (1024,512) bf16, N-major
    const float* __restrict__ rel_bias,       // (1024)
    const float* __restrict__ qkv,            // (T*B, 1536) fp32  [q|k|v]
    unsigned short* __restrict__ attn_out,    // (T*B, 512) bf16
    int T, int B)
{
  __shared__ unsigned short relb[64 * 32];  // bf16 A tile
#if USE_ASYNC
  __shared__ float relf32[2][64 * 32];      // async landing buffers
#endif
  __shared__ float wrow[8 * 64];            // per-head scores / probs

  const int b = blockIdx.x, i = blockIdx.y;
  const int tid = threadIdx.x;
  const int h = tid >> 5;          // head = wave
  const int lane = tid & 31;
  const int l16 = lane & 15, hi = lane >> 4;

  const float* qrow = qkv + (size_t)(i * B + b) * 1536;

  float wacc[32];
#pragma unroll
  for (int t = 0; t < 32; ++t) wacc[t] = 0.f;

#if USE_ASYNC
  auto issue_tile = [&](int ks2, int bufsel) {
#pragma unroll
    for (int ch2 = 0; ch2 < 2; ++ch2) {
      const int ch = tid + ch2 * 256;            // 512 16B chunks / tile
      const int j = ch >> 3, e4 = (ch & 7) * 4;
      // score w[i,j] uses relation[j, i, b, :]  (swapaxes(0,1) in reference)
      const float* g = rel + (((size_t)j * T + i) * B + b) * 512 + ks2 * 32 + e4;
      const unsigned lds_off = (unsigned)(uintptr_t)(&relf32[bufsel][j * 32 + e4]);
      asm volatile("global_load_async_to_lds_b128 %0, %1, off"
                   :: "v"(lds_off), "v"((unsigned long long)(uintptr_t)g)
                   : "memory");
    }
  };
#endif

  for (int c = 0; c < 4; ++c) {     // 16-col sub-chunk within the head
    v8f acc[4][2];
#pragma unroll
    for (int mt = 0; mt < 4; ++mt) { acc[mt][0] = v8f_zero(); acc[mt][1] = v8f_zero(); }

#if USE_ASYNC
    issue_tile(0, 0);
#endif
    for (int ks = 0; ks < 16; ++ks) {
      const int kb = ks * 32;
#if USE_ASYNC
      if (ks + 1 < 16) {
        issue_tile(ks + 1, (ks + 1) & 1);
        asm volatile("s_wait_asynccnt 0x2" ::: "memory");  // tile ks landed
      } else {
        asm volatile("s_wait_asynccnt 0x0" ::: "memory");
      }
      __syncthreads();
      // fp32 -> bf16 in LDS; one ds_load_b64 + one packed ds_store_b32 / pair.
      // 32 lanes x 8B = full 256B bank sweep: conflict-free.
#pragma unroll
      for (int e = 0; e < 4; ++e) {
        const int pr = e * 512 + tid * 2;  // even pair index
        const float2 f2v = *(const float2*)&relf32[ks & 1][pr];
        *(unsigned*)&relb[pr] = pack2_bf16_rte(f2v.x, f2v.y);
      }
      __syncthreads();
#else
      __syncthreads();
#pragma unroll
      for (int ch2 = 0; ch2 < 2; ++ch2) {
        const int ch = tid + ch2 * 256;
        const int j = ch >> 3, e4 = (ch & 7) * 4;
        const float* g = rel + (((size_t)j * T + i) * B + b) * 512 + kb + e4;
        const float4 f = *(const float4*)g;
        const int lo = j * 32 + e4;  // multiple of 4 -> dword aligned
        *(unsigned*)&relb[lo]     = pack2_bf16_rte(f.x, f.y);
        *(unsigned*)&relb[lo + 2] = pack2_bf16_rte(f.z, f.w);
      }
      __syncthreads();
#endif
      // B fragments straight from global bf16 weights (L2 resident, 1MB)
      Frag16 bF[2];
      {
        const int ncol = h * 64 + c * 16 + l16;
        const unsigned short* p0 = relWt + (size_t)ncol * 512 + kb + hi * 16;
        bF[0].q[0] = *(const uint4*)p0;  bF[0].q[1] = *(const uint4*)(p0 + 8);
        const unsigned short* p1 = relWt + (size_t)(512 + ncol) * 512 + kb + hi * 16;
        bF[1].q[0] = *(const uint4*)p1;  bF[1].q[1] = *(const uint4*)(p1 + 8);
      }
#pragma unroll
      for (int mt = 0; mt < 4; ++mt) {
        Frag16 aF;
        const unsigned short* p = &relb[(mt * 16 + l16) * 32 + hi * 16];
        aF.q[0] = *(const uint4*)p;
        aF.q[1] = *(const uint4*)(p + 8);
        acc[mt][0] = wmma_bf16(aF, bF[0], acc[mt][0]);  // ra chunk
        acc[mt][1] = wmma_bf16(aF, bF[1], acc[mt][1]);  // rb chunk
      }
    }

    // fold this chunk into the per-(j,h) score accumulators
    {
      const int ncol = h * 64 + c * 16 + l16;
      const float ba = rel_bias[ncol];
      const float bb = rel_bias[512 + ncol];
      const float qv = qrow[ncol];
#pragma unroll
      for (int mt = 0; mt < 4; ++mt) {
#pragma unroll
        for (int p = 0; p < 8; ++p) {
          const int j = mt * 16 + p + hi * 8;
          const float kv = qkv[(size_t)(j * B + b) * 1536 + 512 + ncol];
          const float ra = acc[mt][0][p] + ba;
          const float rb = acc[mt][1][p] + bb;
          wacc[mt * 8 + p] += (qv + ra) * (kv + rb);
        }
      }
    }
  }

  // reduce over the 16 lanes of each half-wave, write scores
  const float scl = 1.0f / 64.0f;  // (Dh^-0.5)^2
#pragma unroll
  for (int t = 0; t < 32; ++t) {
    float s = wacc[t];
    s += __shfl_xor(s, 1);
    s += __shfl_xor(s, 2);
    s += __shfl_xor(s, 4);
    s += __shfl_xor(s, 8);
    if (l16 == 0) {
      const int j = (t >> 3) * 16 + (t & 7) + hi * 8;
      wrow[h * 64 + j] = s * scl;
    }
  }
  __syncthreads();

  // softmax over j (64 values, 2 per lane) for this head
  const float a0 = wrow[h * 64 + lane];
  const float a1 = wrow[h * 64 + lane + 32];
  float mx = fmaxf(a0, a1);
#pragma unroll
  for (int m = 16; m >= 1; m >>= 1) mx = fmaxf(mx, __shfl_xor(mx, m));
  const float e0 = __expf(a0 - mx), e1 = __expf(a1 - mx);
  float sm = e0 + e1;
#pragma unroll
  for (int m = 16; m >= 1; m >>= 1) sm += __shfl_xor(sm, m);
  const float inv = 1.0f / sm;
  wrow[h * 64 + lane]      = e0 * inv;
  wrow[h * 64 + lane + 32] = e1 * inv;
  __syncthreads();

  // attn[i,b,h,:] = sum_j P[j] * v[j,b,h,:]
  float o0 = 0.f, o1 = 0.f;
  for (int j = 0; j < 64; ++j) {
    const float pj = wrow[h * 64 + j];
    const float* vp = qkv + (size_t)(j * B + b) * 1536 + 1024 + h * 64;
    o0 += pj * vp[lane];
    o1 += pj * vp[lane + 32];
  }
  const size_t orow = (size_t)(i * B + b) * 512 + h * 64;
  attn_out[orow + lane]      = f32_to_bf16_rte(o0);
  attn_out[orow + lane + 32] = f32_to_bf16_rte(o1);
}

// ---------------------------------------------------------------------------
// out = LayerNorm(X + Y) * g + b ; one wave per row (E = 512, 16 elems/lane).
// Optionally also emits a bf16 copy (feeds the next WMMA GEMM).
// ---------------------------------------------------------------------------
__global__ void ln_fused(const float* __restrict__ X, const float* __restrict__ Y,
                         const float* __restrict__ g, const float* __restrict__ be,
                         float* __restrict__ outF, unsigned short* __restrict__ outB)
{
  const int row = blockIdx.x;
  const int lane = threadIdx.x;  // blockDim 32
  const float* x = X + (size_t)row * 512;
  const float* y = Y + (size_t)row * 512;
  float v[16];
  float s = 0.f;
#pragma unroll
  for (int t = 0; t < 16; ++t) { v[t] = x[lane + t * 32] + y[lane + t * 32]; s += v[t]; }
#pragma unroll
  for (int m = 16; m >= 1; m >>= 1) s += __shfl_xor(s, m);
  const float mu = s * (1.0f / 512.0f);
  float var = 0.f;
#pragma unroll
  for (int t = 0; t < 16; ++t) { const float d = v[t] - mu; var += d * d; }
#pragma unroll
  for (int m = 16; m >= 1; m >>= 1) var += __shfl_xor(var, m);
  const float inv = rsqrtf(var * (1.0f / 512.0f) + 1e-5f);
#pragma unroll
  for (int t = 0; t < 16; ++t) {
    const int n = lane + t * 32;
    const float o = (v[t] - mu) * inv * g[n] + be[n];
    if (outF) outF[(size_t)row * 512 + n] = o;
    if (outB) outB[(size_t)row * 512 + n] = f32_to_bf16_rte(o);
  }
}

// --------------------------- conversion helpers ----------------------------
__global__ void cvt_flat(const float* __restrict__ in, unsigned short* __restrict__ out,
                         int n) {
  const int idx = blockIdx.x * blockDim.x + threadIdx.x;
  if (idx < n) out[idx] = f32_to_bf16_rte(in[idx]);
}
// out[c*R + r] = bf16(in[r*C + c])   (produce N-major weights)
__global__ void cvt_transpose(const float* __restrict__ in,
                              unsigned short* __restrict__ out, int R, int C) {
  const int idx = blockIdx.x * blockDim.x + threadIdx.x;
  if (idx < R * C) {
    const int r = idx / C, c = idx % C;
    out[(size_t)c * R + r] = f32_to_bf16_rte(in[idx]);
  }
}

// ---------------------------------------------------------------------------
extern "C" void kernel_launch(void* const* d_in, const int* in_sizes, int n_in,
                              void* d_out, int out_size, void* d_ws, size_t ws_size,
                              hipStream_t stream)
{
  const float* x     = (const float*)d_in[0];
  const float* rel   = (const float*)d_in[1];
  const float* in_w  = (const float*)d_in[2];
  const float* in_b  = (const float*)d_in[3];
  const float* rel_w = (const float*)d_in[4];
  const float* rel_b = (const float*)d_in[5];
  const float* out_w = (const float*)d_in[6];
  const float* out_b = (const float*)d_in[7];
  const float* fc1_w = (const float*)d_in[8];
  const float* fc1_b = (const float*)d_in[9];
  const float* fc2_w = (const float*)d_in[10];
  const float* fc2_b = (const float*)d_in[11];
  const float* ln1_g = (const float*)d_in[12];
  const float* ln1_b = (const float*)d_in[13];
  const float* ln2_g = (const float*)d_in[14];
  const float* ln2_b = (const float*)d_in[15];

  const int E = 512, F = 2048;
  const int T = in_sizes[1] / in_sizes[0];     // relation/x element ratio
  const int B = in_sizes[0] / (T * E);
  const int M = T * B;                          // 2048 token-rows

  char* ws = (char*)d_ws;
  size_t off = 0;
  auto alloc = [&](size_t bytes) -> void* {
    void* p = ws + off;
    off += (bytes + 255) & ~(size_t)255;
    return p;
  };
  unsigned short* x_bf    = (unsigned short*)alloc((size_t)M * E * 2);
  unsigned short* inw_bf  = (unsigned short*)alloc((size_t)3 * E * E * 2);
  unsigned short* relWt   = (unsigned short*)alloc((size_t)2 * E * E * 2);
  unsigned short* outWt   = (unsigned short*)alloc((size_t)E * E * 2);
  unsigned short* fc1Wt   = (unsigned short*)alloc((size_t)E * F * 2);
  unsigned short* fc2Wt   = (unsigned short*)alloc((size_t)E * F * 2);
  float*          qkv     = (float*)alloc((size_t)M * 3 * E * 4);
  unsigned short* attn_bf = (unsigned short*)alloc((size_t)M * E * 2);
  float*          aproj   = (float*)alloc((size_t)M * E * 4);
  float*          h_f     = (float*)alloc((size_t)M * E * 4);
  unsigned short* h_bf    = (unsigned short*)alloc((size_t)M * E * 2);
  unsigned short* f1_bf   = (unsigned short*)alloc((size_t)M * F * 2);
  float*          f2      = (float*)alloc((size_t)M * E * 4);
  (void)ws_size; (void)n_in; (void)out_size;

  const int CT = 256;
  auto g1 = [&](int n) { return dim3((unsigned)((n + CT - 1) / CT)); };

  // Weight / activation precision prep (one-time, tiny vs 256MB relation read)
  cvt_flat<<<g1(M * E), CT, 0, stream>>>(x, x_bf, M * E);
  cvt_flat<<<g1(3 * E * E), CT, 0, stream>>>(in_w, inw_bf, 3 * E * E);  // already N-major
  cvt_transpose<<<g1(2 * E * E), CT, 0, stream>>>(rel_w, relWt, E, 2 * E);
  cvt_transpose<<<g1(E * E), CT, 0, stream>>>(out_w, outWt, E, E);
  cvt_transpose<<<g1(E * F), CT, 0, stream>>>(fc1_w, fc1Wt, E, F);
  cvt_transpose<<<g1(E * F), CT, 0, stream>>>(fc2_w, fc2Wt, F, E);

  // qkv = x @ in_w^T + in_b
  gemm_bf16<0><<<dim3(3 * E / 128, M / 128), 256, 0, stream>>>(
      x_bf, inw_bf, in_b, qkv, nullptr, M, 3 * E, E);

  // fused relation projection + scores + softmax + P@V  (dominant kernel)
  rel_attn<<<dim3(B, T), 256, 0, stream>>>(rel, relWt, rel_b, qkv, attn_bf, T, B);

  // out projection, residual + LN1 (emits fp32 h and bf16 h)
  gemm_bf16<0><<<dim3(E / 128, M / 128), 256, 0, stream>>>(
      attn_bf, outWt, out_b, aproj, nullptr, M, E, E);
  ln_fused<<<M, 32, 0, stream>>>(x, aproj, ln1_g, ln1_b, h_f, h_bf);

  // FFN: relu(h@fc1 + b1) @ fc2 + b2, residual + LN2 -> d_out
  gemm_bf16<1><<<dim3(F / 128, M / 128), 256, 0, stream>>>(
      h_bf, fc1Wt, fc1_b, nullptr, f1_bf, M, F, E);
  gemm_bf16<0><<<dim3(E / 128, M / 128), 256, 0, stream>>>(
      f1_bf, fc2Wt, fc2_b, f2, nullptr, M, E, F);
  ln_fused<<<M, 32, 0, stream>>>(h_f, f2, ln2_g, ln2_b, (float*)d_out, nullptr);
}
